// NeuralRDE_55396488184014
// MI455X (gfx1250) — matrix operand order repaired and checked by reading, not compiled, not run on hardware
//
#include <hip/hip_runtime.h>
#include <hip/hip_bf16.h>
#include <cstddef>

// Problem constants (match reference)
#define B_   512
#define NI_  256
#define LS_  64
#define D_   8
#define H_   128
#define WID  256
#define LAB_ 10
#define LL   63       // LS - 1
#define NTOT (H_*LL)  // 8064
#define NSTEPS_ 256

typedef __attribute__((ext_vector_type(16))) __bf16 bf16x16;
typedef __attribute__((ext_vector_type(8)))  __bf16 bf16x8;
typedef __attribute__((ext_vector_type(8)))  float  f32x8;

// ---------------------------------------------------------------- utilities

__global__ void f32_to_bf16_k(const float* __restrict__ src,
                              __bf16* __restrict__ dst, int n) {
  int i = blockIdx.x * blockDim.x + threadIdx.x;
  if (i < n) dst[i] = (__bf16)src[i];
}

// y0 = x0 @ l1w^T + l1b   (K=8, too small for WMMA)
__global__ void init_y0_k(const float* __restrict__ x0, const float* __restrict__ l1w,
                          const float* __restrict__ l1b, float* __restrict__ y,
                          __bf16* __restrict__ ybf) {
  int i = blockIdx.x * blockDim.x + threadIdx.x;
  if (i >= B_ * H_) return;
  int b = i / H_, d = i % H_;
  float s = l1b[d];
#pragma unroll
  for (int k = 0; k < D_; k++) s += x0[b * D_ + k] * l1w[d * D_ + k];
  y[i] = s;
  ybf[i] = (__bf16)s;
}

// yout = y + dt*(c1*k1 + ... + c6*k6); writes bf16 (stage input) and optional f32 (state)
__global__ void combine_k(const float* __restrict__ y,
                          const float* __restrict__ k1, const float* __restrict__ k2,
                          const float* __restrict__ k3, const float* __restrict__ k4,
                          const float* __restrict__ k5, const float* __restrict__ k6,
                          float c1, float c2, float c3, float c4, float c5, float c6,
                          float dt, float* __restrict__ yf32, __bf16* __restrict__ ybf,
                          int n) {
  int i = blockIdx.x * blockDim.x + threadIdx.x;
  if (i >= n) return;
  float a = 0.f;
  if (c1 != 0.f) a += c1 * k1[i];
  if (c2 != 0.f) a += c2 * k2[i];
  if (c3 != 0.f) a += c3 * k3[i];
  if (c4 != 0.f) a += c4 * k4[i];
  if (c5 != 0.f) a += c5 * k5[i];
  if (c6 != 0.f) a += c6 * k6[i];
  float v = y[i] + dt * a;
  if (yf32) yf32[i] = v;
  ybf[i] = (__bf16)v;
}

// ------------------------------------------------------------------------
// Shared tile helpers: B (weight) tile 64(N) x 32(K) staged in LDS,
// double-buffered; one 32-byte chunk per thread (128 threads).
// LDS layout: row-major [n][k], so a B fragment (K akb..akb+7 and akb+16..akb+23 of one n)
// is two 16-byte LDS reads.
// ------------------------------------------------------------------------

__device__ __forceinline__ void stage_B(__bf16* dst, const __bf16* __restrict__ Wt,
                                        int nrow, int K, int kk, int shalf) {
  *(bf16x16*)(dst) = *(const bf16x16*)(Wt + (size_t)nrow * K + kk + shalf);
}

// Build A fragment for 16 rows starting at arow (per-lane row), K-step kk.
__device__ __forceinline__ bf16x16 load_Afrag(const __bf16* arow, int kk, int akb) {
  bf16x8 alo = *(const bf16x8*)(arow + kk + akb);
  bf16x8 ahi = *(const bf16x8*)(arow + kk + akb + 16);
  bf16x16 af;
#pragma unroll
  for (int i = 0; i < 8; i++) { af[i] = alo[i]; af[8 + i] = ahi[i]; }
  return af;
}

// ---------------------------------------------------------------- WMMA GEMM + silu
// out[m,n] = silu( sum_k A[m,k]*Wt[n,k] + bias[n] ), A: MxK bf16, Wt: NxK bf16.
// Block = 128 threads (4 waves); block tile 128(M) x 64(N); wave tile 32x64.
__global__ __launch_bounds__(128)
void gemm_silu_k(const __bf16* __restrict__ A, const __bf16* __restrict__ Wt,
                 const float* __restrict__ bias, __bf16* __restrict__ out,
                 int M, int N, int K) {
  __shared__ __bf16 sB[2][64 * 32];

  const int tid  = threadIdx.x;
  const int lane = tid & 31;
  const int wave = tid >> 5;
  const int half = lane >> 4;     // 0: lanes 0-15, 1: lanes 16-31
  const int l16  = lane & 15;
  const int m0 = blockIdx.x * 128 + wave * 32;
  const int n0 = blockIdx.y * 64;

  const int srow  = tid >> 1;          // 0..63  (staging row)
  const int shalf = (tid & 1) * 16;    // staging k-half

  f32x8 acc[2][4] = {};
  const __bf16* arow0 = A + (size_t)(m0 + l16) * K;
  const __bf16* arow1 = A + (size_t)(m0 + 16 + l16) * K;
  const int akb = half ? 8 : 0;    // A fragment K-base within 32-step
  const int bkb = half ? 16 : 0;   // unused; B fragment uses akb too

  // prime buffer 0
  stage_B(&sB[0][srow * 32 + shalf], Wt, n0 + srow, K, 0, shalf);
  __syncthreads();

  for (int kk = 0; kk < K; kk += 32) {
    const int p = (kk >> 5) & 1;
    if (kk + 32 < K)
      stage_B(&sB[p ^ 1][srow * 32 + shalf], Wt, n0 + srow, K, kk + 32, shalf);

    bf16x16 af0 = load_Afrag(arow0, kk, akb);
    bf16x16 af1 = load_Afrag(arow1, kk, akb);
#pragma unroll
    for (int t = 0; t < 4; t++) {
      bf16x16 bfr = load_Afrag(&sB[p][(16 * t + l16) * 32], 0, akb);
      acc[0][t] = __builtin_amdgcn_wmma_f32_16x16x32_bf16(
          false, af0, false, bfr, (short)0, acc[0][t], false, false);
      acc[1][t] = __builtin_amdgcn_wmma_f32_16x16x32_bf16(
          false, af1, false, bfr, (short)0, acc[1][t], false, false);
    }
    __syncthreads();
  }

#pragma unroll
  for (int t = 0; t < 4; t++) {
    int col = n0 + 16 * t + l16;
    float bv = bias[col];
#pragma unroll
    for (int mt = 0; mt < 2; mt++) {
#pragma unroll
      for (int j = 0; j < 8; j++) {
        int row = m0 + mt * 16 + j + 8 * half;
        float v = acc[mt][t][j] + bv;
        float s = v / (1.f + __expf(-v));            // silu
        out[(size_t)row * N + col] = (__bf16)s;
      }
    }
  }
}

// ---------------------------------------------------------------- WMMA GEMM + tanh + einsum
// kout[b,h] = sum_{l<63} tanh( (h2 @ W2^T)[b, h*63+l] + b2[h*63+l] ) * logsig[b, idx, 1+l]
// Grid: (M/128, H). Block 128 threads; wave tile 32(b) x 64(l; col 63 masked).
__global__ __launch_bounds__(128)
void gemm_tanh_reduce_k(const __bf16* __restrict__ A,       // B_ x 256 bf16 (h2)
                        const __bf16* __restrict__ W2b,     // 8064 x 256 bf16
                        const float* __restrict__ b2,
                        const float* __restrict__ logsig,   // B_ x NI_ x LS_ f32
                        const float* __restrict__ intervals,
                        float tcur, float* __restrict__ kout) {
  __shared__ __bf16 sB[2][64 * 32];

  const int K = WID;
  const int tid  = threadIdx.x;
  const int lane = tid & 31;
  const int wave = tid >> 5;
  const int half = lane >> 4;
  const int l16  = lane & 15;
  const int m0 = blockIdx.x * 128 + wave * 32;
  const int h  = blockIdx.y;
  const int nbase = h * LL;

  const int srow  = tid >> 1;
  const int shalf = (tid & 1) * 16;
  int nstage = nbase + srow;
  if (nstage >= NTOT) nstage = NTOT - 1;  // clamp (masked in reduction)

  int ncol[4];
#pragma unroll
  for (int t = 0; t < 4; t++) {
    int n = nbase + 16 * t + l16;
    ncol[t] = (n < NTOT) ? n : (NTOT - 1);
  }

  f32x8 acc[2][4] = {};
  const __bf16* arow0 = A + (size_t)(m0 + l16) * K;
  const __bf16* arow1 = A + (size_t)(m0 + 16 + l16) * K;
  const int akb = half ? 8 : 0;
  const int bkb = half ? 16 : 0;

  stage_B(&sB[0][srow * 32 + shalf], W2b, nstage, K, 0, shalf);
  __syncthreads();

  for (int kk = 0; kk < K; kk += 32) {
    const int p = (kk >> 5) & 1;
    if (kk + 32 < K)
      stage_B(&sB[p ^ 1][srow * 32 + shalf], W2b, nstage, K, kk + 32, shalf);

    bf16x16 af0 = load_Afrag(arow0, kk, akb);
    bf16x16 af1 = load_Afrag(arow1, kk, akb);
#pragma unroll
    for (int t = 0; t < 4; t++) {
      bf16x16 bfr = load_Afrag(&sB[p][(16 * t + l16) * 32], 0, akb);
      acc[0][t] = __builtin_amdgcn_wmma_f32_16x16x32_bf16(
          false, af0, false, bfr, (short)0, acc[0][t], false, false);
      acc[1][t] = __builtin_amdgcn_wmma_f32_16x16x32_bf16(
          false, af1, false, bfr, (short)0, acc[1][t], false, false);
    }
    __syncthreads();
  }

  // idx = clip(searchsorted(intervals, t, side='left'), 0, NI-1)
  int lo = 0, hi = NI_;
  while (lo < hi) { int mid = (lo + hi) >> 1; if (intervals[mid] < tcur) lo = mid + 1; else hi = mid; }
  const int idx = (lo < NI_ - 1) ? lo : (NI_ - 1);

  float b2v[4];
#pragma unroll
  for (int t = 0; t < 4; t++) b2v[t] = b2[ncol[t]];

#pragma unroll
  for (int mt = 0; mt < 2; mt++) {
#pragma unroll
    for (int j = 0; j < 8; j++) {
      int row = m0 + mt * 16 + j + 8 * half;
      const float* grow = logsig + (size_t)row * NI_ * LS_ + (size_t)idx * LS_ + 1;
      float p = 0.f;
#pragma unroll
      for (int t = 0; t < 4; t++) {
        int c = 16 * t + l16;
        float g = (c < LL) ? grow[c] : 0.f;
        p += tanhf(acc[mt][t][j] + b2v[t]) * g;
      }
      // reduce across the 16 lanes of this half-wave (columns)
      p += __shfl_xor(p, 1, 32);
      p += __shfl_xor(p, 2, 32);
      p += __shfl_xor(p, 4, 32);
      p += __shfl_xor(p, 8, 32);
      if (l16 == 0) kout[(size_t)row * H_ + h] = p;
    }
  }
}

// ---------------------------------------------------------------- head + softmax
__global__ void head_softmax_k(const float* __restrict__ y, const float* __restrict__ l2w,
                               const float* __restrict__ l2b, float* __restrict__ out) {
  int b = blockIdx.x * blockDim.x + threadIdx.x;
  if (b >= B_) return;
  float logit[LAB_];
  float mx = -1e30f;
#pragma unroll
  for (int n = 0; n < LAB_; n++) {
    float s = l2b[n];
    for (int k = 0; k < H_; k++) s += y[b * H_ + k] * l2w[n * H_ + k];
    logit[n] = s;
    mx = s > mx ? s : mx;
  }
  float den = 0.f;
#pragma unroll
  for (int n = 0; n < LAB_; n++) { logit[n] = __expf(logit[n] - mx); den += logit[n]; }
#pragma unroll
  for (int n = 0; n < LAB_; n++) out[b * LAB_ + n] = logit[n] / den;
}

// ---------------------------------------------------------------- driver

extern "C" void kernel_launch(void* const* d_in, const int* in_sizes, int n_in,
                              void* d_out, int out_size, void* d_ws, size_t ws_size,
                              hipStream_t stream) {
  (void)in_sizes; (void)n_in; (void)out_size; (void)ws_size;
  const float* intervals = (const float*)d_in[1];
  const float* logsig    = (const float*)d_in[2];
  const float* x0        = (const float*)d_in[3];
  const float* W0        = (const float*)d_in[4];
  const float* b0        = (const float*)d_in[5];
  const float* W1        = (const float*)d_in[6];
  const float* b1        = (const float*)d_in[7];
  const float* W2        = (const float*)d_in[8];
  const float* b2        = (const float*)d_in[9];
  const float* l1w       = (const float*)d_in[10];
  const float* l1b       = (const float*)d_in[11];
  const float* l2w       = (const float*)d_in[12];
  const float* l2b       = (const float*)d_in[13];

  char* ws = (char*)d_ws;
  size_t off = 0;
  auto alloc = [&](size_t bytes) -> void* {
    void* p = ws + off;
    off += (bytes + 255) & ~(size_t)255;
    return p;
  };
  __bf16* bW0  = (__bf16*)alloc((size_t)WID * H_ * 2);
  __bf16* bW1  = (__bf16*)alloc((size_t)WID * WID * 2);
  __bf16* bW2  = (__bf16*)alloc((size_t)NTOT * WID * 2);
  float*  y    = (float*) alloc((size_t)B_ * H_ * 4);
  __bf16* ybf  = (__bf16*)alloc((size_t)B_ * H_ * 2);
  __bf16* ytmp = (__bf16*)alloc((size_t)B_ * H_ * 2);
  __bf16* h1   = (__bf16*)alloc((size_t)B_ * WID * 2);
  __bf16* h2   = (__bf16*)alloc((size_t)B_ * WID * 2);
  float* kk[6];
  for (int i = 0; i < 6; i++) kk[i] = (float*)alloc((size_t)B_ * H_ * 4);

  // bf16 weight copies (every call; deterministic)
  f32_to_bf16_k<<<(WID * H_ + 255) / 256, 256, 0, stream>>>(W0, bW0, WID * H_);
  f32_to_bf16_k<<<(WID * WID + 255) / 256, 256, 0, stream>>>(W1, bW1, WID * WID);
  f32_to_bf16_k<<<(NTOT * WID + 255) / 256, 256, 0, stream>>>(W2, bW2, NTOT * WID);
  init_y0_k<<<(B_ * H_ + 255) / 256, 256, 0, stream>>>(x0, l1w, l1b, y, ybf);

  const float dt = 1.0f / 256.0f;  // (ts[-1]-ts[0])/NSTEPS with ts=linspace(0,1,257)
  const float Cs[6] = {0.f, 0.161f, 0.327f, 0.9f, 0.9800255409045097f, 1.0f};
  const float Ac[5][5] = {
      {0.161f, 0.f, 0.f, 0.f, 0.f},
      {-0.008480655492356989f, 0.335480655492357f, 0.f, 0.f, 0.f},
      {2.8971530571054935f, -6.359448489975075f, 4.3622954328695815f, 0.f, 0.f},
      {5.325864828439257f, -11.748883564062828f, 7.4955393428898365f,
       -0.09249506636175525f, 0.f},
      {5.86145544294642f, -12.92096931784711f, 8.159367898576159f,
       -0.071584973281401f, -0.028269050394068383f}};
  const float Bc[6] = {0.09646076681806523f, 0.01f, 0.4798896504144996f,
                       1.379008574103742f, -3.290069515436081f, 2.324710524099774f};

  const dim3 blk(128);
  const dim3 g1(B_ / 128, WID / 64);  // GEMM1/GEMM2 grids (4 x 4)
  const dim3 g3(B_ / 128, H_);        // W2 fused grid (4 x 128)
  const int nY = B_ * H_;

  float t = 0.f;
  for (int step = 0; step < NSTEPS_; step++) {
    for (int s = 0; s < 6; s++) {
      const __bf16* yin;
      if (s == 0) {
        yin = ybf;
      } else {
        const float* a = Ac[s - 1];
        combine_k<<<(nY + 255) / 256, 256, 0, stream>>>(
            y, kk[0], kk[1], kk[2], kk[3], kk[4], kk[5],
            a[0], a[1], a[2], a[3], a[4], 0.f, dt, (float*)nullptr, ytmp, nY);
        yin = ytmp;
      }
      gemm_silu_k<<<g1, blk, 0, stream>>>(yin, bW0, b0, h1, B_, WID, H_);
      gemm_silu_k<<<g1, blk, 0, stream>>>(h1, bW1, b1, h2, B_, WID, WID);
      gemm_tanh_reduce_k<<<g3, blk, 0, stream>>>(h2, bW2, b2, logsig, intervals,
                                                 t + Cs[s] * dt, kk[s]);
    }
    combine_k<<<(nY + 255) / 256, 256, 0, stream>>>(
        y, kk[0], kk[1], kk[2], kk[3], kk[4], kk[5],
        Bc[0], Bc[1], Bc[2], Bc[3], Bc[4], Bc[5], dt, y, ybf, nY);
    t += dt;
  }

  head_softmax_k<<<(B_ + 63) / 64, 64, 0, stream>>>(y, l2w, l2b, (float*)d_out);
}
